// MultiHeadSelfAttention_25881472926142
// MI455X (gfx1250) — matrix operand order repaired and checked by reading, not compiled, not run on hardware
//
#include <hip/hip_runtime.h>

// ---------------------------------------------------------------------------
// Types for CDNA5 WMMA (wave32, v_wmma_f32_16x16x32_bf16)
// ---------------------------------------------------------------------------
typedef __bf16 bf16_t;
typedef __attribute__((ext_vector_type(16))) __bf16 v16bf;
typedef __attribute__((ext_vector_type(8)))  __bf16 v8bf;
typedef __attribute__((ext_vector_type(8)))  float  v8f;

#define WMMA_BF16(A, B, C) \
  __builtin_amdgcn_wmma_f32_16x16x32_bf16(false, (A), false, (B), (short)0, (C), false, false)

#define QK_SCALE 0.08838834764831845f  // 1/sqrt(128)

__device__ __forceinline__ v16bf cat8(v8bf lo, v8bf hi) {
  return __builtin_shufflevector(lo, hi, 0, 1, 2, 3, 4, 5, 6, 7,
                                         8, 9, 10, 11, 12, 13, 14, 15);
}

// A-fragment (16x32 bf16, rows M x contraction K) from row-major [rows, ld]:
// lane: row = l&15, K = 16*(l>>4) .. +15 contiguous
__device__ __forceinline__ v16bf load_fragA_nt(const bf16_t* base, int row0, int k0, int ld) {
  int lane = threadIdx.x & 31;
  int half = lane >> 4;
  int r    = lane & 15;
  const bf16_t* p = base + (size_t)(row0 + r) * ld + k0 + half * 16;
  return cat8(*(const v8bf*)p, *(const v8bf*)(p + 8));
}

// B-fragment (contraction K x cols N) where memory is row-major [N, K]:
// lane: col = l&15, K = 16*(l>>4) .. +15 contiguous
__device__ __forceinline__ v16bf load_fragB_nt(const bf16_t* base, int n0, int k0, int ld) {
  int lane = threadIdx.x & 31;
  int half = lane >> 4;
  int c    = lane & 15;
  const bf16_t* p = base + (size_t)(n0 + c) * ld + k0 + half * 16;
  return cat8(*(const v8bf*)p, *(const v8bf*)(p + 8));
}

// ---------------------------------------------------------------------------
// f32 -> bf16 conversion
// ---------------------------------------------------------------------------
__global__ __launch_bounds__(256) void f32_to_bf16_kernel(const float* __restrict__ in,
                                                          bf16_t* __restrict__ out, int n) {
  int i = blockIdx.x * 256 + threadIdx.x;
  if (i < n) out[i] = (bf16_t)in[i];
}

// ---------------------------------------------------------------------------
// NT GEMM: C[M,N] = A[M,K] * B[N,K]^T   (bf16 in, f32 accumulate), pipelined.
// block = 256 threads = 8 waves (2 M x 4 N), wave tile = 32x32 (2x2 WMMA).
// MODE 0: f32 output (ldc)            -> Cf
// MODE 2: QKV output:
//   n <  2048 : q, scaled by 1/sqrt(128), bf16 -> Cb [2048,4096]
//   n <  4096 : k, bf16                       -> Cb [2048,4096]
//   n >= 4096 : v, stored TRANSPOSED bf16     -> Cv [2048,2048]  (Cv[d_glob][t])
// ---------------------------------------------------------------------------
template <int MODE>
__global__ __launch_bounds__(256) void gemm_nt_wmma(const bf16_t* __restrict__ A,
                                                    const bf16_t* __restrict__ B,
                                                    float* __restrict__ Cf,
                                                    bf16_t* __restrict__ Cb,
                                                    bf16_t* __restrict__ Cv,
                                                    int K, int lda, int ldb, int ldc) {
  int wave  = threadIdx.x >> 5;
  int waveM = wave >> 2;     // 0..1
  int waveN = wave & 3;      // 0..3
  int m0 = blockIdx.x * 64 + waveM * 32;
  int n0 = blockIdx.y * 128 + waveN * 32;
  int lane = threadIdx.x & 31;

  v8f acc00 = {}, acc01 = {}, acc10 = {}, acc11 = {};

  v16bf a0 = load_fragA_nt(A, m0,      0, lda);
  v16bf a1 = load_fragA_nt(A, m0 + 16, 0, lda);
  v16bf b0 = load_fragB_nt(B, n0,      0, ldb);
  v16bf b1 = load_fragB_nt(B, n0 + 16, 0, ldb);

  for (int k0 = 0; k0 < K; k0 += 32) {
    v16bf ca0 = a0, ca1 = a1, cb0 = b0, cb1 = b1;
    int kn = k0 + 32;
    if (kn < K) {  // software pipeline: issue next tile's loads before WMMAs
      a0 = load_fragA_nt(A, m0,      kn, lda);
      a1 = load_fragA_nt(A, m0 + 16, kn, lda);
      b0 = load_fragB_nt(B, n0,      kn, ldb);
      b1 = load_fragB_nt(B, n0 + 16, kn, ldb);
    }
    if (kn + 96 < K) {  // global_prefetch_b8, three k-tiles past kn
      __builtin_prefetch(A + (size_t)(m0 + lane) * lda + kn + 96, 0, 0);
      __builtin_prefetch(B + (size_t)(n0 + lane) * ldb + kn + 96, 0, 0);
    }
    acc00 = WMMA_BF16(ca0, cb0, acc00);
    acc01 = WMMA_BF16(ca0, cb1, acc01);
    acc10 = WMMA_BF16(ca1, cb0, acc10);
    acc11 = WMMA_BF16(ca1, cb1, acc11);
  }

  int col  = lane & 15;
  int rowb = (lane >> 4) * 8;

  if (MODE == 2 && n0 >= 4096) {
    // V portion: store transposed, vectorized (lane owns 8 consecutive t rows)
    v8bf r00, r01, r10, r11;
#pragma unroll
    for (int r = 0; r < 8; ++r) {
      r00[r] = (bf16_t)acc00[r];
      r01[r] = (bf16_t)acc01[r];
      r10[r] = (bf16_t)acc10[r];
      r11[r] = (bf16_t)acc11[r];
    }
    size_t nA = (size_t)(n0 - 4096 + col);
    size_t nB = nA + 16;
    *(v8bf*)(Cv + nA * 2048 + m0 + rowb)      = r00;
    *(v8bf*)(Cv + nB * 2048 + m0 + rowb)      = r01;
    *(v8bf*)(Cv + nA * 2048 + m0 + 16 + rowb) = r10;
    *(v8bf*)(Cv + nB * 2048 + m0 + 16 + rowb) = r11;
    return;
  }

  float s = (MODE == 2 && n0 < 2048) ? QK_SCALE : 1.0f;
#pragma unroll
  for (int r = 0; r < 8; ++r) {
    int rA = m0 + rowb + r, rB = m0 + 16 + rowb + r;
    if (MODE == 0) {
      Cf[(size_t)rA * ldc + n0 + col]      = acc00[r];
      Cf[(size_t)rA * ldc + n0 + 16 + col] = acc01[r];
      Cf[(size_t)rB * ldc + n0 + col]      = acc10[r];
      Cf[(size_t)rB * ldc + n0 + 16 + col] = acc11[r];
    } else {
      Cb[(size_t)rA * ldc + n0 + col]      = (bf16_t)(acc00[r] * s);
      Cb[(size_t)rA * ldc + n0 + 16 + col] = (bf16_t)(acc01[r] * s);
      Cb[(size_t)rB * ldc + n0 + col]      = (bf16_t)(acc10[r] * s);
      Cb[(size_t)rB * ldc + n0 + 16 + col] = (bf16_t)(acc11[r] * s);
    }
  }
}

// ---------------------------------------------------------------------------
// Fused flash attention (per head), bf16 WMMA, online softmax.
// qk : bf16 [2048, 4096]  (cols: q_scaled | k), per head = 128 cols
// vT : bf16 [2048, 2048]  (row = h*128 + d, col = t)  -- pre-transposed V
// ctx: bf16 [2048, 2048]
// grid = (32, 16) blocks of 128 threads (4 waves); wave handles 16 query rows.
// V^T tiles staged into LDS with global_load_async_to_lds_b128 (ASYNCcnt).
// ---------------------------------------------------------------------------
#define TLEN  2048
#define QK_LD 4096

__global__ __launch_bounds__(128) void attn_flash_wmma(const bf16_t* __restrict__ qk,
                                                       const bf16_t* __restrict__ vT,
                                                       bf16_t* __restrict__ ctx) {
  __shared__ __align__(16) bf16_t vt[128 * 32];     // V^T tile: [d][s], 8 KB
  __shared__ __align__(16) bf16_t pbuf[4][16 * 32]; // per-wave P tile, 4 KB

  const int h    = blockIdx.y;
  const int tid  = threadIdx.x;
  const int wave = tid >> 5;
  const int lane = tid & 31;
  const int half = lane >> 4;
  const int r16  = lane & 15;
  const int t0   = blockIdx.x * 64 + wave * 16;

  const bf16_t* qb  = qk + (size_t)h * 128;
  const bf16_t* kb  = qk + 2048 + (size_t)h * 128;
  const bf16_t* vtg = vT + (size_t)h * 128 * 2048;
  const unsigned vtbase = (unsigned)(unsigned long long)(&vt[0]);

  // Q fragments (q already scaled by 1/sqrt(128) at QKV-GEMM store)
  v16bf qf[4];
#pragma unroll
  for (int c = 0; c < 4; ++c) qf[c] = load_fragA_nt(qb, t0, 32 * c, QK_LD);

  v8f o[8] = {};              // O accumulator 16x128 in C layout
  float mstate[8], lstate[8]; // online softmax state (replicated per 16-lane half)
#pragma unroll
  for (int r = 0; r < 8; ++r) { mstate[r] = -3.0e38f; lstate[r] = 0.0f; }

  for (int s0 = 0; s0 < TLEN; s0 += 32) {
    __syncthreads(); // previous iteration's vt reads complete

    // async-copy V^T tile: vT[h*128+d][s0..s0+31] -> vt[d*32 + s]  (8 KB)
    // both sides contiguous in 16B chunks; 4 chunks per thread
#pragma unroll
    for (int i = tid; i < 512; i += 128) {
      int d   = i >> 2;
      int sc8 = (i & 3) * 8;
      const bf16_t* g = vtg + (size_t)d * 2048 + s0 + sc8;
      unsigned ldsoff = vtbase + (unsigned)(i * 16);
      asm volatile("global_load_async_to_lds_b128 %0, %1, off"
                   :: "v"(ldsoff), "v"(g) : "memory");
    }

    // K fragments for this step (overlap the async copy in flight)
    v16bf kf[8];
#pragma unroll
    for (int c = 0; c < 4; ++c) {
      kf[c]     = load_fragB_nt(kb, s0,      32 * c, QK_LD);
      kf[4 + c] = load_fragB_nt(kb, s0 + 16, 32 * c, QK_LD);
    }

    // S tile = Qs (16x128) . K^T -> 16x32 scores (two 16x16 C tiles)
    v8f sc0 = {}, sc1 = {};
#pragma unroll
    for (int c = 0; c < 4; ++c) sc0 = WMMA_BF16(qf[c], kf[c], sc0);
#pragma unroll
    for (int c = 0; c < 4; ++c) sc1 = WMMA_BF16(qf[c], kf[4 + c], sc1);

    // online softmax; row = r + 8*half, cols spread across 16 lanes of half
    bf16_t* pb = &pbuf[wave][0];
#pragma unroll
    for (int r = 0; r < 8; ++r) {
      float a0 = sc0[r];
      float a1 = sc1[r];
      float rm = fmaxf(a0, a1);
      rm = fmaxf(rm, __shfl_xor(rm, 1, 32));
      rm = fmaxf(rm, __shfl_xor(rm, 2, 32));
      rm = fmaxf(rm, __shfl_xor(rm, 4, 32));
      rm = fmaxf(rm, __shfl_xor(rm, 8, 32));
      float mnew  = fmaxf(mstate[r], rm);
      float alpha = __expf(mstate[r] - mnew);
      float p0 = __expf(a0 - mnew);
      float p1 = __expf(a1 - mnew);
      float rs = p0 + p1;
      rs += __shfl_xor(rs, 1, 32);
      rs += __shfl_xor(rs, 2, 32);
      rs += __shfl_xor(rs, 4, 32);
      rs += __shfl_xor(rs, 8, 32);
      lstate[r] = lstate[r] * alpha + rs;
      mstate[r] = mnew;
#pragma unroll
      for (int d = 0; d < 8; ++d) o[d][r] *= alpha;
      int row = 8 * half + r;
      pb[row * 32 + r16]      = (bf16_t)p0;
      pb[row * 32 + 16 + r16] = (bf16_t)p1;
    }
    asm volatile("s_wait_dscnt 0x0" ::: "memory");

    // reread P as WMMA A-fragment (16x32)
    const bf16_t* pr = pb + (size_t)r16 * 32 + half * 16;
    v16bf pf = cat8(*(const v8bf*)pr, *(const v8bf*)(pr + 8));

    // V^T tile must be resident before B-fragment reads
    asm volatile("s_wait_asynccnt 0x0" ::: "memory");
    __syncthreads();

    // O += P (16x32) . V (32x128): B-fragments from vt (contiguous along s)
#pragma unroll
    for (int d = 0; d < 8; ++d) {
      const bf16_t* vp = vt + (size_t)(16 * d + r16) * 32 + half * 16;
      v16bf vf = cat8(*(const v8bf*)vp, *(const v8bf*)(vp + 8));
      o[d] = WMMA_BF16(pf, vf, o[d]);
    }
  }

  // finalize: O * (1/l), write ctx[t, h*128 + d] as bf16
#pragma unroll
  for (int r = 0; r < 8; ++r) {
    float rl = __builtin_amdgcn_rcpf(lstate[r]);
#pragma unroll
    for (int d = 0; d < 8; ++d) {
      float val = o[d][r] * rl;
      ctx[(size_t)(t0 + 8 * half + r) * 2048 + h * 128 + 16 * d + r16] = (bf16_t)val;
    }
  }
}

// ---------------------------------------------------------------------------
// Launch: x[2048,2048]f32, w_in[6144,2048]f32, w_out[2048,2048]f32 -> out f32
// ---------------------------------------------------------------------------
extern "C" void kernel_launch(void* const* d_in, const int* in_sizes, int n_in,
                              void* d_out, int out_size, void* d_ws, size_t ws_size,
                              hipStream_t stream) {
  const float* x     = (const float*)d_in[0];
  const float* w_in  = (const float*)d_in[1];
  const float* w_out = (const float*)d_in[2];
  float* out = (float*)d_out;

  char* ws = (char*)d_ws;
  bf16_t* xb      = (bf16_t*)(ws);             // [2048,2048]        8 MB
  bf16_t* w_in_b  = (bf16_t*)(ws + 8388608);   // [6144,2048]       24 MB
  bf16_t* w_out_b = (bf16_t*)(ws + 33554432);  // [2048,2048]        8 MB
  bf16_t* qk_b    = (bf16_t*)(ws + 41943040);  // [2048,4096] q|k   16 MB
  bf16_t* vT_b    = (bf16_t*)(ws + 58720256);  // [2048,2048] V^T    8 MB
  bf16_t* ctx_b   = (bf16_t*)(ws + 67108864);  // [2048,2048]        8 MB

  // f32 -> bf16
  {
    int n;
    n = 2048 * 2048;
    f32_to_bf16_kernel<<<(n + 255) / 256, 256, 0, stream>>>(x, xb, n);
    n = 6144 * 2048;
    f32_to_bf16_kernel<<<(n + 255) / 256, 256, 0, stream>>>(w_in, w_in_b, n);
    n = 2048 * 2048;
    f32_to_bf16_kernel<<<(n + 255) / 256, 256, 0, stream>>>(w_out, w_out_b, n);
  }

  // qkv = x @ w_in^T : q (scaled) | k -> qk_b, v -> vT_b transposed
  gemm_nt_wmma<2><<<dim3(32, 48), 256, 0, stream>>>(
      xb, w_in_b, nullptr, qk_b, vT_b, /*K=*/2048, 2048, 2048, /*ldc=*/4096);

  // fused multi-head attention -> ctx [2048, 2048] bf16
  attn_flash_wmma<<<dim3(32, 16), 128, 0, stream>>>(qk_b, vT_b, ctx_b);

  // out = ctx @ w_out^T : [2048, 2048] f32
  gemm_nt_wmma<0><<<dim3(32, 16), 256, 0, stream>>>(
      ctx_b, w_out_b, out, nullptr, nullptr, /*K=*/2048, 2048, 2048, /*ldc=*/2048);
}
